// CostVolume_6511170421288
// MI455X (gfx1250) — matrix operand order repaired and checked by reading, not compiled, not run on hardware
//
#include <hip/hip_runtime.h>

// CostVolume correlation via banded GEMM on WMMA f16 (gfx1250 / CDNA5).
// cost[b,d,h,w] = (1/C) sum_c L[b,c,h,w] * R[b,c,h,w-d],  0 for w<d.
//
// Per (b,h): band 0 <= w_l - w_r < 48 of L(WxC) * R(WxC)^T, K=128.
// HBM-bound (~560 MB total traffic at CHUNK=160 => ~24 us at 23.3 TB/s);
// WMMA f16 makes the 6.7 GFLOP reduction negligible.

#define BATCH    8
#define CHAN     128
#define HEIGHT   160
#define WIDTH    320
#define MAXD     48
#define CHUNK    160             // w-columns per workgroup (2 chunks per row)
#define HALO     48              // extra right-feature columns to the left
#define RWID     (CHUNK + HALO)  // 208
#define LPAD     8               // pad rows to 136 halves (272B, 16B aligned)
#define LSTR     (CHAN + LPAD)
#define OSTR     (CHUNK + 1)     // output staging row stride (floats)
#define NTHREADS 320             // 10 waves of 32, one 16-row tile per wave

#define L_HALVES (CHUNK * LSTR)                    // 21760
#define R_HALVES (RWID * LSTR)                     // 28288
#define SMEM_BYTES ((L_HALVES + R_HALVES) * 2)     // 100096 B (ldsO aliases ldsL)

typedef __attribute__((ext_vector_type(16))) _Float16 v16h;
typedef __attribute__((ext_vector_type(8)))  _Float16 v8h;
typedef __attribute__((ext_vector_type(2)))  _Float16 h2;
typedef __attribute__((ext_vector_type(8)))  float    v8f;

__global__ __launch_bounds__(NTHREADS)
void cost_volume_wmma(const float* __restrict__ left,
                      const float* __restrict__ right,
                      float* __restrict__ out) {
    extern __shared__ __align__(16) char smem[];
    _Float16* ldsL = (_Float16*)smem;              // [CHUNK][LSTR]
    _Float16* ldsR = ldsL + L_HALVES;              // [RWID][LSTR]
    float*    ldsO = (float*)smem;                 // [MAXD][OSTR], aliases ldsL

    const int t  = threadIdx.x;
    const int w0 = blockIdx.x * CHUNK;
    const int h  = blockIdx.y;
    const int b  = blockIdx.z;

    const size_t planeHW = (size_t)HEIGHT * WIDTH;
    const float* Lbase = left  + (size_t)b * CHAN * planeHW + (size_t)h * WIDTH;
    const float* Rbase = right + (size_t)b * CHAN * planeHW + (size_t)h * WIDTH;

    // ---- Stage left: 160(w) x 128(c) f32 -> f16, transposed to LDS[w][c].
    // Each thread: two channel rows (c,c+1) x 4 w => packed b32 LDS stores.
    #pragma unroll
    for (int k = 0; k < (CHUNK / 4) * (CHAN / 2) / NTHREADS; ++k) {   // 8
        int idx = k * NTHREADS + t;
        int cp  = idx / (CHUNK / 4);          // channel pair 0..63
        int w4  = (idx % (CHUNK / 4)) * 4;    // 0..156
        const float4 v0 = *(const float4*)(Lbase + (size_t)(2 * cp)     * planeHW + w0 + w4);
        const float4 v1 = *(const float4*)(Lbase + (size_t)(2 * cp + 1) * planeHW + w0 + w4);
        *(h2*)(ldsL + (w4 + 0) * LSTR + 2 * cp) = h2{(_Float16)v0.x, (_Float16)v1.x};
        *(h2*)(ldsL + (w4 + 1) * LSTR + 2 * cp) = h2{(_Float16)v0.y, (_Float16)v1.y};
        *(h2*)(ldsL + (w4 + 2) * LSTR + 2 * cp) = h2{(_Float16)v0.z, (_Float16)v1.z};
        *(h2*)(ldsL + (w4 + 3) * LSTR + 2 * cp) = h2{(_Float16)v0.w, (_Float16)v1.w};
    }
    // ---- Stage right: 208(w) x 128(c), halo of 48 to the left, zero-filled
    // for global w < 0 (implements the reference's zero-padding for free).
    #pragma unroll
    for (int k = 0; k < ((RWID / 4) * (CHAN / 2) + NTHREADS - 1) / NTHREADS; ++k) { // 11
        int idx = k * NTHREADS + t;
        if (idx < (RWID / 4) * (CHAN / 2)) {
            int cp  = idx / (RWID / 4);           // channel pair 0..63
            int w4  = (idx % (RWID / 4)) * 4;     // 0..204
            int wg  = w0 - HALO + w4;             // 4-aligned global w
            float4 v0, v1;
            if (wg >= 0) {
                v0 = *(const float4*)(Rbase + (size_t)(2 * cp)     * planeHW + wg);
                v1 = *(const float4*)(Rbase + (size_t)(2 * cp + 1) * planeHW + wg);
            } else {
                v0 = make_float4(0.f, 0.f, 0.f, 0.f);
                v1 = v0;
            }
            *(h2*)(ldsR + (w4 + 0) * LSTR + 2 * cp) = h2{(_Float16)v0.x, (_Float16)v1.x};
            *(h2*)(ldsR + (w4 + 1) * LSTR + 2 * cp) = h2{(_Float16)v0.y, (_Float16)v1.y};
            *(h2*)(ldsR + (w4 + 2) * LSTR + 2 * cp) = h2{(_Float16)v0.z, (_Float16)v1.z};
            *(h2*)(ldsR + (w4 + 3) * LSTR + 2 * cp) = h2{(_Float16)v0.w, (_Float16)v1.w};
        }
    }
    __syncthreads();

    // ---- Banded GEMM: wave `it` owns output rows w_l in [w0+16*it, +16).
    // Column tiles Delta=0..3 give d = 16*Delta + m - n covering d in [0,63].
    const int it   = t >> 5;          // row tile, 0..9
    const int lane = t & 31;
    const int mrow = lane & 15;       // matrix row (A) / column (B) this lane holds
    const int hl   = lane >> 4;       // lane half selects K sub-block

    v8f acc[4] = {};
    #pragma unroll
    for (int kc = 0; kc < CHAN; kc += 32) {
        // A fragment (16x32 f16): lane half holds K {0..7,16..23} / {8..15,24..31}
        const _Float16* arow = ldsL + (16 * it + mrow) * LSTR;
        v8h a0 = *(const v8h*)(arow + kc + 8 * hl);
        v8h a1 = *(const v8h*)(arow + kc + 16 + 8 * hl);
        v16h a = __builtin_shufflevector(a0, a1,
                 0, 1, 2, 3, 4, 5, 6, 7, 8, 9, 10, 11, 12, 13, 14, 15);
        #pragma unroll
        for (int dl = 0; dl < 4; ++dl) {
            // B fragment (32x16 f16): lane = column (w_r), lane half holds
            // K {0..7,16..23} / {8..15,24..31}
            const _Float16* brow = ldsR + (HALO + 16 * (it - dl) + mrow) * LSTR;
            v8h b0 = *(const v8h*)(brow + kc + 8 * hl);
            v8h b1 = *(const v8h*)(brow + kc + 16 + 8 * hl);
            v16h bb = __builtin_shufflevector(b0, b1,
                      0, 1, 2, 3, 4, 5, 6, 7, 8, 9, 10, 11, 12, 13, 14, 15);
            acc[dl] = __builtin_amdgcn_wmma_f32_16x16x32_f16(
                false, a, false, bb, (short)0, acc[dl], false, false);
        }
    }
    __syncthreads();   // ldsO aliases ldsL: all fragment reads must be done

    // ---- Scatter D tiles to (d, w) staging. Each valid (d,w) is written
    // exactly once across the 4 Delta tiles.
    const float invC = 1.0f / (float)CHAN;
    #pragma unroll
    for (int dl = 0; dl < 4; ++dl) {
        #pragma unroll
        for (int v = 0; v < 8; ++v) {
            int m = v + 8 * hl;            // row within tile -> w offset
            int d = 16 * dl + m - mrow;    // disparity
            if (d >= 0 && d < MAXD)
                ldsO[d * OSTR + 16 * it + m] = acc[dl][v] * invC;
        }
    }
    __syncthreads();

    // ---- Coalesced store: 48 rows x 160 contiguous floats (640B each).
    #pragma unroll
    for (int k = 0; k < MAXD * CHUNK / NTHREADS; ++k) {    // 24
        int idx = k * NTHREADS + t;
        int d   = idx / CHUNK;
        int w   = idx % CHUNK;
        out[(((size_t)b * MAXD + d) * HEIGHT + h) * WIDTH + w0 + w] = ldsO[d * OSTR + w];
    }
}

extern "C" void kernel_launch(void* const* d_in, const int* in_sizes, int n_in,
                              void* d_out, int out_size, void* d_ws, size_t ws_size,
                              hipStream_t stream) {
    const float* left  = (const float*)d_in[0];
    const float* right = (const float*)d_in[1];
    float* out = (float*)d_out;
    // Opt in to >64KB dynamic LDS (CDNA5 WGP has 320KB). Not a stream op;
    // safe under graph capture and deterministic per call.
    (void)hipFuncSetAttribute(reinterpret_cast<const void*>(cost_volume_wmma),
                              hipFuncAttributeMaxDynamicSharedMemorySize,
                              SMEM_BYTES);
    dim3 grid(WIDTH / CHUNK, HEIGHT, BATCH);
    cost_volume_wmma<<<grid, NTHREADS, SMEM_BYTES, stream>>>(left, right, out);
}
